// LinearMultiheadAttention_67534065762953
// MI455X (gfx1250) — matrix operand order repaired and checked by reading, not compiled, not run on hardware
//
#include <hip/hip_runtime.h>
#include <cstdint>

// ---------------------------------------------------------------------------
// Problem constants (LinearMultiheadAttention reference)
// ---------------------------------------------------------------------------
constexpr int Lg  = 4096;
constexpr int Bg  = 4;
constexpr int Dg  = 1024;
constexpr int Hg  = 16;
constexpr int Pg  = 256;
constexpr int HDg = 64;          // head dim
constexpr float SCAL = 0.125f;   // HD^-0.5

typedef __attribute__((ext_vector_type(16))) __bf16 v16bf;
typedef __attribute__((ext_vector_type(8)))  float  v8f;
typedef int v4i __attribute__((ext_vector_type(4)));

// gfx1250 async global->LDS path (guarded: falls back to sync staging)
#if defined(__AMDGCN__) && __has_builtin(__builtin_amdgcn_global_load_async_to_lds_b128)
#define HAVE_ASYNC 1
#else
#define HAVE_ASYNC 0
#endif

__device__ __forceinline__ void wait_async0() {
#if __has_builtin(__builtin_amdgcn_s_wait_asynccnt)
  __builtin_amdgcn_s_wait_asynccnt(0);
#elif defined(__AMDGCN__)
  asm volatile("s_wait_asynccnt 0x0" ::: "memory");
#endif
}

// ---------------------------------------------------------------------------
// One-time f32 -> bf16 conversion (n must be a multiple of 1024; all are)
// ---------------------------------------------------------------------------
__global__ __launch_bounds__(256) void f32_to_bf16_kernel(
    const float* __restrict__ src, __bf16* __restrict__ dst, long n) {
  const long i = ((long)blockIdx.x * 256 + threadIdx.x) * 4;
  if (i + 3 < n) {
    const float4 v = *(const float4*)(src + i);
    dst[i + 0] = (__bf16)v.x; dst[i + 1] = (__bf16)v.y;
    dst[i + 2] = (__bf16)v.z; dst[i + 3] = (__bf16)v.w;
  }
}

// ---------------------------------------------------------------------------
// Batched bf16 WMMA GEMM:  C[m,n] = epi( sum_k A[m,k]*B[n,k] )
//   A element: Ab[m*sAm + k]         (K always contiguous in A)
//   B element: Bb[n*sBn + k*sBk]     (STAGE_B => sBk==1, tile staged in LDS)
//   batch z = i1*nz2 + i2 with per-dim strides for A/B/C.
//   epi: r = (acc + bias[n]) * alpha; if (relu) r = max(r,0)
// Wave tile 16(M) x 64(N); block = 4 waves -> 64x64; B tile shared via LDS.
// ---------------------------------------------------------------------------
template <typename TC, bool STAGE_B>
__global__ __launch_bounds__(128) void gemm_wmma_kernel(
    const __bf16* __restrict__ A, const __bf16* __restrict__ Bm, TC* __restrict__ C,
    const float* __restrict__ bias,
    long sA1, long sA2, long sAm,
    long sB1, long sB2, long sBn, long sBk,
    long sC1, long sC2, long ldc,
    int nz2, int K, float alpha, int relu)
{
  const int tid  = threadIdx.x;
  const int lane = tid & 31, wave = tid >> 5;
  const int half = lane >> 4, lr = lane & 15;

  const int i1 = blockIdx.z / nz2, i2 = blockIdx.z % nz2;
  const __bf16* Ab = A  + i1 * sA1 + i2 * sA2;
  const __bf16* Bb = Bm + i1 * sB1 + i2 * sB2;
  TC*           Cb = C  + i1 * sC1 + i2 * sC2;

  const long m  = (long)blockIdx.y * 64 + wave * 16 + lr;
  const long n0 = (long)blockIdx.x * 64;

  // A-matrix 16x32 bf16 lane layout: lanes 0-15 K 0-15, lanes 16-31 K 16-31 (matches B)
  int koffA[16];
#pragma unroll
  for (int i = 0; i < 16; ++i) koffA[i] = i + half * 16;
  // B-matrix 32x16 lane layout: lanes 0-15 K 0-15, lanes 16-31 K 16-31
  const int kb0 = half * 16;

  v8f acc[4] = {};

  if constexpr (STAGE_B) {
    // double-buffered 64x32 bf16 B tile (4 KB per buffer)
    __shared__ __attribute__((aligned(32))) __bf16 bt[2][64][32];
    auto stage = [&](int kb, int buf) {
#pragma unroll
      for (int j = 0; j < 2; ++j) {
        const int c   = tid + j * 128;           // 256 x 16B chunks
        const int row = c >> 2, ks = (c & 3) * 8;
        const __bf16* gp = Bb + (long)(n0 + row) * sBn + kb + ks;
#if HAVE_ASYNC
        __builtin_amdgcn_global_load_async_to_lds_b128(
            (__attribute__((address_space(1))) v4i*)(uintptr_t)gp,
            (__attribute__((address_space(3))) v4i*)(uint32_t)(uintptr_t)&bt[buf][row][ks],
            0, 0);
#else
        *(uint4*)&bt[buf][row][ks] = *(const uint4*)gp;
#endif
      }
    };

    stage(0, 0);
    int buf = 0;
    for (int kb = 0; kb < K; kb += 32) {
#if HAVE_ASYNC
      wait_async0();            // own async writes done
#endif
      __syncthreads();          // everyone's writes visible; prev buffer free
      if (kb + 32 < K) stage(kb + 32, buf ^ 1);

      const __bf16* ap = Ab + m * sAm + kb;
      if (kb + 32 < K) __builtin_prefetch(ap + 32, 0, 0);
      v16bf a;
#pragma unroll
      for (int i = 0; i < 16; ++i) a[i] = ap[koffA[i]];
#pragma unroll
      for (int nt = 0; nt < 4; ++nt) {
        const v16bf bfr = *(const v16bf*)&bt[buf][nt * 16 + lr][kb0];
        acc[nt] = __builtin_amdgcn_wmma_f32_16x16x32_bf16(
            false, a, false, bfr, (short)0, acc[nt], false, false);
      }
      buf ^= 1;
    }
  } else {
    // direct path for strided-K B operands (stage-3 contraction)
    for (int kb = 0; kb < K; kb += 32) {
      const __bf16* ap = Ab + m * sAm + kb;
      v16bf a;
#pragma unroll
      for (int i = 0; i < 16; ++i) a[i] = ap[koffA[i]];
#pragma unroll
      for (int nt = 0; nt < 4; ++nt) {
        const long n = n0 + nt * 16 + lr;
        v16bf bfr;
#pragma unroll
        for (int i = 0; i < 16; ++i)
          bfr[i] = Bb[n * sBn + (long)(kb + kb0 + i) * sBk];
        acc[nt] = __builtin_amdgcn_wmma_f32_16x16x32_bf16(
            false, a, false, bfr, (short)0, acc[nt], false, false);
      }
    }
  }

#pragma unroll
  for (int nt = 0; nt < 4; ++nt) {
    const long n = n0 + nt * 16 + lr;
    const float bv = bias ? bias[n] : 0.0f;
#pragma unroll
    for (int r = 0; r < 8; ++r) {
      float v = (acc[nt][r] + bv) * alpha;
      if (relu) v = fmaxf(v, 0.0f);
      const long mm = (long)blockIdx.y * 64 + wave * 16 + r + 8 * half;
      Cb[mm * ldc + n] = (TC)v;
    }
  }
}

// ---------------------------------------------------------------------------
// LayerNorm(eo + e_weight residual) over D=1024, block (256 thr) per row.
// Writes bf16 (consumed only by K/V WMMA projections).
// ---------------------------------------------------------------------------
__global__ __launch_bounds__(256) void ln_res_kernel(
    const float* __restrict__ eo, const float* __restrict__ ew,
    const float* __restrict__ g,  const float* __restrict__ bta,
    __bf16* __restrict__ out)
{
  const int row = blockIdx.x;      // p*B + b
  const int p   = row >> 2;        // B = 4
  const int t   = threadIdx.x;
  __shared__ float red[256];

  float x[4];
  float s = 0.0f;
#pragma unroll
  for (int j = 0; j < 4; ++j) {
    const int d = t + j * 256;
    const int h = d >> 6, hd = d & 63;
    x[j] = eo[(size_t)row * Dg + d] + ew[(size_t)(h * Pg + p) * HDg + hd];
    s += x[j];
  }
  red[t] = s; __syncthreads();
  for (int off = 128; off > 0; off >>= 1) {
    if (t < off) red[t] += red[t + off];
    __syncthreads();
  }
  const float mean = red[0] * (1.0f / Dg);
  __syncthreads();
  s = 0.0f;
#pragma unroll
  for (int j = 0; j < 4; ++j) { const float d0 = x[j] - mean; s += d0 * d0; }
  red[t] = s; __syncthreads();
  for (int off = 128; off > 0; off >>= 1) {
    if (t < off) red[t] += red[t + off];
    __syncthreads();
  }
  const float inv = rsqrtf(red[0] * (1.0f / Dg) + 1e-5f);
#pragma unroll
  for (int j = 0; j < 4; ++j) {
    const int d = t + j * 256;
    out[(size_t)row * Dg + d] = (__bf16)((x[j] - mean) * inv * g[d] + bta[d]);
  }
}

// ---------------------------------------------------------------------------
// Fused attention: one wave = 16 query rows of one (b,h).
//   logits(16x256) -> LDS; softmax over P=256 on-chip; probs @ V -> bf16.
// ---------------------------------------------------------------------------
__global__ __launch_bounds__(32) void attn_kernel(
    const __bf16* __restrict__ qbf, const __bf16* __restrict__ kbf,
    const __bf16* __restrict__ vbf, __bf16* __restrict__ attnb)
{
  __shared__ float  logits[16][Pg];   // 16 KB
  __shared__ __bf16 probs [16][Pg];   //  8 KB
  const int lane = threadIdx.x;
  const int half = lane >> 4, lr = lane & 15;
  const int h  = blockIdx.y;
  const int bb = blockIdx.z;
  const int l0 = blockIdx.x * 16;

  int koffA[16];
#pragma unroll
  for (int i = 0; i < 16; ++i) koffA[i] = i + half * 16;
  const int kb0 = half * 16;

  const __bf16* qrow = qbf + ((size_t)(l0 + lr) * Bg + bb) * Dg + h * HDg;
  v16bf a0, a1;
#pragma unroll
  for (int i = 0; i < 16; ++i) { a0[i] = qrow[koffA[i]]; a1[i] = qrow[32 + koffA[i]]; }

  // ---- logits = qh @ kh^T (16 x 256) ----
#pragma unroll 4
  for (int pt = 0; pt < 16; ++pt) {
    const __bf16* krow = kbf + ((size_t)(pt * 16 + lr) * Bg + bb) * Dg + h * HDg;
    const v16bf b0 = *(const v16bf*)(krow + kb0);        // 32B contiguous
    const v16bf b1 = *(const v16bf*)(krow + 32 + kb0);
    v8f c = {};
    c = __builtin_amdgcn_wmma_f32_16x16x32_bf16(false, a0, false, b0, (short)0, c, false, false);
    c = __builtin_amdgcn_wmma_f32_16x16x32_bf16(false, a1, false, b1, (short)0, c, false, false);
#pragma unroll
    for (int r = 0; r < 8; ++r) logits[r + 8 * half][pt * 16 + lr] = c[r];
  }

  // ---- softmax over P: lanes 0..15 each own one row ----
  if (lane < 16) {
    float mx = -1e30f;
    for (int pp = 0; pp < Pg; ++pp) mx = fmaxf(mx, logits[lane][pp]);
    float ssum = 0.0f;
    for (int pp = 0; pp < Pg; ++pp) {
      const float e = __expf(logits[lane][pp] - mx);
      logits[lane][pp] = e;
      ssum += e;
    }
    const float inv = 1.0f / ssum;
    for (int pp = 0; pp < Pg; ++pp) probs[lane][pp] = (__bf16)(logits[lane][pp] * inv);
  }

  // ---- attn = probs(16x256) @ V(256x64) ----
  v8f o[4] = {};
#pragma unroll 2
  for (int kt = 0; kt < 8; ++kt) {
    v16bf pa;
#pragma unroll
    for (int i = 0; i < 16; ++i) pa[i] = probs[lr][kt * 32 + koffA[i]];
#pragma unroll
    for (int nt = 0; nt < 4; ++nt) {
      v16bf vf;
#pragma unroll
      for (int i = 0; i < 16; ++i)
        vf[i] = vbf[((size_t)(kt * 32 + kb0 + i) * Bg + bb) * Dg + h * HDg + nt * 16 + lr];
      o[nt] = __builtin_amdgcn_wmma_f32_16x16x32_bf16(
          false, pa, false, vf, (short)0, o[nt], false, false);
    }
  }
#pragma unroll
  for (int nt = 0; nt < 4; ++nt)
#pragma unroll
    for (int r = 0; r < 8; ++r)
      attnb[((size_t)(l0 + r + 8 * half) * Bg + bb) * Dg + h * HDg + nt * 16 + lr] =
          (__bf16)o[nt][r];
}

// ---------------------------------------------------------------------------
// Host launcher
// ---------------------------------------------------------------------------
extern "C" void kernel_launch(void* const* d_in, const int* in_sizes, int n_in,
                              void* d_out, int out_size, void* d_ws, size_t ws_size,
                              hipStream_t stream) {
  (void)in_sizes; (void)n_in; (void)out_size; (void)ws_size;
  const float* query = (const float*)d_in[0];
  const float* q_w   = (const float*)d_in[1];
  const float* q_b   = (const float*)d_in[2];
  const float* k_w   = (const float*)d_in[3];
  const float* k_b   = (const float*)d_in[4];
  const float* v_w   = (const float*)d_in[5];
  const float* v_b   = (const float*)d_in[6];
  const float* e_w   = (const float*)d_in[7];   // (1,H,P,HD)
  const float* eo_w  = (const float*)d_in[8];
  const float* eo_b  = (const float*)d_in[9];
  const float* ln_g  = (const float*)d_in[10];
  const float* ln_b  = (const float*)d_in[11];
  const float* out_w = (const float*)d_in[12];
  const float* out_b = (const float*)d_in[13];
  float* outp = (float*)d_out;

  char* ws = (char*)d_ws;
  auto carve = [&](size_t bytes) -> char* {
    char* pp = ws; ws += (bytes + 255) & ~(size_t)255; return pp;
  };
  __bf16* qry_bf  = (__bf16*)carve((size_t)Lg * Bg * Dg * 2);      //  32 MB
  __bf16* qw_bf   = (__bf16*)carve((size_t)Dg * Dg * 2);           //   2 MB
  __bf16* kw_bf   = (__bf16*)carve((size_t)Dg * Dg * 2);
  __bf16* vw_bf   = (__bf16*)carve((size_t)Dg * Dg * 2);
  __bf16* eow_bf  = (__bf16*)carve((size_t)Dg * Dg * 2);
  __bf16* ow_bf   = (__bf16*)carve((size_t)Dg * Dg * 2);
  __bf16* ew_bf   = (__bf16*)carve((size_t)Hg * Pg * HDg * 2);     // 0.5 MB
  __bf16* q_bf    = (__bf16*)carve((size_t)Lg * Bg * Dg * 2);      //  32 MB
  __bf16* pkv     = (__bf16*)carve((size_t)Hg * Bg * Pg * Lg * 2); // 128 MB [h][b][p][l]
  __bf16* kvraw   = (__bf16*)carve((size_t)Pg * Bg * Dg * 2);      //   2 MB (P*B, D)
  float*  eo      = (float*) carve((size_t)Pg * Bg * Dg * 4);      //   4 MB
  __bf16* kvln_bf = (__bf16*)carve((size_t)Pg * Bg * Dg * 2);      //   2 MB
  __bf16* k_bf    = (__bf16*)carve((size_t)Pg * Bg * Dg * 2);
  __bf16* v_bf    = (__bf16*)carve((size_t)Pg * Bg * Dg * 2);
  __bf16* at_bf   = (__bf16*)carve((size_t)Lg * Bg * Dg * 2);      //  32 MB

  const dim3 blk(128);

  // 0) one-time bf16 conversions
  f32_to_bf16_kernel<<<(Lg*Bg*Dg)/1024, 256, 0, stream>>>(query, qry_bf, (long)Lg*Bg*Dg);
  f32_to_bf16_kernel<<<(Dg*Dg)/1024, 256, 0, stream>>>(q_w,  qw_bf,  (long)Dg*Dg);
  f32_to_bf16_kernel<<<(Dg*Dg)/1024, 256, 0, stream>>>(k_w,  kw_bf,  (long)Dg*Dg);
  f32_to_bf16_kernel<<<(Dg*Dg)/1024, 256, 0, stream>>>(v_w,  vw_bf,  (long)Dg*Dg);
  f32_to_bf16_kernel<<<(Dg*Dg)/1024, 256, 0, stream>>>(eo_w, eow_bf, (long)Dg*Dg);
  f32_to_bf16_kernel<<<(Dg*Dg)/1024, 256, 0, stream>>>(out_w, ow_bf, (long)Dg*Dg);
  f32_to_bf16_kernel<<<(Hg*Pg*HDg)/1024, 256, 0, stream>>>(e_w, ew_bf, (long)Hg*Pg*HDg);

  // 1) q = (query @ q_w^T + q_b) * scaling    (L*B, D) -> bf16
  gemm_wmma_kernel<__bf16, true><<<dim3(Dg/64, (Lg*Bg)/64, 1), blk, 0, stream>>>(
      qry_bf, qw_bf, q_bf, q_b,
      0, 0, Dg,   0, 0, Dg, 1,   0, 0, Dg,
      1, Dg, SCAL, 0);

  // 2) pkv[h][b] = relu(scaling * E_h @ kvh)   M=P, N=L, K=HD, z=h*B+b
  gemm_wmma_kernel<__bf16, true><<<dim3(Lg/64, Pg/64, Hg*Bg), blk, 0, stream>>>(
      ew_bf, qry_bf, pkv, nullptr,
      (long)Pg*HDg, 0, HDg,
      HDg, Dg, (long)Bg*Dg, 1,
      (long)Bg*Pg*Lg, (long)Pg*Lg, Lg,
      Bg, HDg, SCAL, 1);

  // 3) kv_raw = pkv[h][b] @ kvh^T   M=P, N=HD, K=L  (strided-K B -> direct)
  gemm_wmma_kernel<__bf16, false><<<dim3(HDg/64, Pg/64, Hg*Bg), blk, 0, stream>>>(
      pkv, qry_bf, kvraw, nullptr,
      (long)Bg*Pg*Lg, (long)Pg*Lg, Lg,
      HDg, Dg, 1, (long)Bg*Dg,
      HDg, Dg, (long)Bg*Dg,
      Bg, Lg, 1.0f, 0);

  // 4) eo = kv_raw @ e_out_w^T + e_out_b      (P*B, D) f32
  gemm_wmma_kernel<float, true><<<dim3(Dg/64, (Pg*Bg)/64, 1), blk, 0, stream>>>(
      kvraw, eow_bf, eo, eo_b,
      0, 0, Dg,   0, 0, Dg, 1,   0, 0, Dg,
      1, Dg, 1.0f, 0);

  // 5) kv_ln = LayerNorm(eo + e_weight residual) -> bf16
  ln_res_kernel<<<dim3(Pg*Bg), dim3(256), 0, stream>>>(eo, e_w, ln_g, ln_b, kvln_bf);

  // 6/7) k, v projections -> bf16
  gemm_wmma_kernel<__bf16, true><<<dim3(Dg/64, (Pg*Bg)/64, 1), blk, 0, stream>>>(
      kvln_bf, kw_bf, k_bf, k_b,
      0, 0, Dg,   0, 0, Dg, 1,   0, 0, Dg,
      1, Dg, 1.0f, 0);
  gemm_wmma_kernel<__bf16, true><<<dim3(Dg/64, (Pg*Bg)/64, 1), blk, 0, stream>>>(
      kvln_bf, vw_bf, v_bf, v_b,
      0, 0, Dg,   0, 0, Dg, 1,   0, 0, Dg,
      1, Dg, 1.0f, 0);

  // 8) fused softmax attention -> attn bf16 (L,B,D)
  attn_kernel<<<dim3(Lg/16, Hg, Bg), dim3(32), 0, stream>>>(q_bf, k_bf, v_bf, at_bf);

  // 9) out = attn @ out_w^T + out_b -> d_out (f32)
  gemm_wmma_kernel<float, true><<<dim3(Dg/64, (Lg*Bg)/64, 1), blk, 0, stream>>>(
      at_bf, ow_bf, outp, out_b,
      0, 0, Dg,   0, 0, Dg, 1,   0, 0, Dg,
      1, Dg, 1.0f, 0);
}
